// PointNetPlusPlusSeg_86165633892452
// MI455X (gfx1250) — matrix operand order repaired and checked by reading, not compiled, not run on hardware
//
#include <hip/hip_runtime.h>

// ---------------------------------------------------------------------------
// PointNet++ part-segmentation forward pass for MI455X (gfx1250, wave32).
//
// v3: 16x64 macro-tile GEMM (4 N-tiles per wave).
//  - A-fragment loaded once per K-step, shared across 4 independent WMMA
//    accumulation chains -> 2.5 b128 loads per v_wmma, 4x less A traffic,
//    latency hidden by independent chains.
//  - Weights device-packed to transposed K-padded f16 (one 32B load per
//    B-fragment); activations f16 with physical zero-padding (guard-free).
// ---------------------------------------------------------------------------

typedef __attribute__((ext_vector_type(16))) _Float16 v16h;
typedef __attribute__((ext_vector_type(8)))  _Float16 v8h;
typedef __attribute__((ext_vector_type(8)))  float    v8f;

struct LayerP {
    const _Float16* Wp;  // packed [C, Kp] transposed f16, zero-padded
    const float* b;      // [C] bias
    const float* g;      // gamma
    const float* bt;     // beta
    const float* mu;     // running mean
    const float* var;    // running var
    int Kp;              // ceil32(K)
};
struct Mlp3 { LayerP l[3]; };

#define BN_EPS 1e-5f

// ---------------------------------------------------------------------------
// Weight pack: Wp[c*Kp + k] = (k < K) ? (f16)W[k*C + c] : 0
// ---------------------------------------------------------------------------
__global__ void pack_w_kernel(const float* __restrict__ W, int K, int C, int Kp,
                              _Float16* __restrict__ Wp)
{
    const int gId = blockIdx.x * blockDim.x + threadIdx.x;
    if (gId >= C * Kp) return;
    const int c = gId / Kp, k = gId % Kp;
    Wp[gId] = (k < K) ? (_Float16)W[(size_t)k * C + c] : (_Float16)0.f;
}

// ---------------------------------------------------------------------------
// Farthest point sampling. One block per batch, seed = Nsrc/2, records the
// pre-update "far" index each iteration (matches the reference scan).
// ---------------------------------------------------------------------------
__global__ void __launch_bounds__(256)
fps_kernel(const float* __restrict__ xyz, int Nsrc, int npoint,
           float* __restrict__ newXyz /* [B, npoint, 3] */)
{
    const int b   = blockIdx.x;
    const int tid = threadIdx.x;
    const int T   = blockDim.x;

    __shared__ float dist[4096];
    __shared__ float rv[256];
    __shared__ int   ri[256];
    __shared__ float cent[3];
    __shared__ int   curFar;

    const float* P = xyz + (size_t)b * Nsrc * 3;

    for (int i = tid; i < Nsrc; i += T) dist[i] = 3.4e38f;
    if (tid == 0) curFar = Nsrc / 2;
    __syncthreads();

    for (int it = 0; it < npoint; ++it) {
        const int far = curFar;
        if (tid < 3) {
            float v = P[(size_t)far * 3 + tid];
            cent[tid] = v;
            newXyz[((size_t)b * npoint + it) * 3 + tid] = v;
        }
        __syncthreads();

        float best = -1.0f; int bi = 0;
        for (int i = tid; i < Nsrc; i += T) {
            float dx = P[i * 3 + 0] - cent[0];
            float dy = P[i * 3 + 1] - cent[1];
            float dz = P[i * 3 + 2] - cent[2];
            float d  = dx * dx + dy * dy + dz * dz;
            float nd = fminf(dist[i], d);
            dist[i] = nd;
            if (nd > best) { best = nd; bi = i; }
        }
        rv[tid] = best; ri[tid] = bi;
        __syncthreads();
        for (int s = T >> 1; s > 0; s >>= 1) {
            if (tid < s) {
                if (rv[tid + s] > rv[tid] ||
                    (rv[tid + s] == rv[tid] && ri[tid + s] < ri[tid])) {
                    rv[tid] = rv[tid + s]; ri[tid] = ri[tid + s];
                }
            }
            __syncthreads();
        }
        if (tid == 0) curFar = ri[0];
        __syncthreads();
    }
}

// ---------------------------------------------------------------------------
// kNN (ascending distance, ties -> lower index). One block per query point.
// ---------------------------------------------------------------------------
__global__ void __launch_bounds__(128)
knn_kernel(const float* __restrict__ qXyz, const float* __restrict__ srcXyz,
           int S, int Nsrc, int k,
           int* __restrict__ outIdx, float* __restrict__ outDist)
{
    const int bs  = blockIdx.x;          // b*S + s
    const int b   = bs / S;
    const int tid = threadIdx.x;
    const int T   = blockDim.x;

    __shared__ float dist[4096];
    __shared__ float rv[128];
    __shared__ int   ri[128];
    __shared__ float q[3];

    if (tid < 3) q[tid] = qXyz[(size_t)bs * 3 + tid];
    __syncthreads();

    const float* P = srcXyz + (size_t)b * Nsrc * 3;
    for (int i = tid; i < Nsrc; i += T) {
        float dx = P[i * 3 + 0] - q[0];
        float dy = P[i * 3 + 1] - q[1];
        float dz = P[i * 3 + 2] - q[2];
        dist[i] = dx * dx + dy * dy + dz * dz;
    }
    __syncthreads();

    for (int j = 0; j < k; ++j) {
        float best = 3.8e38f; int bi = 0;
        for (int i = tid; i < Nsrc; i += T) {
            float v = dist[i];
            if (v < best) { best = v; bi = i; }
        }
        rv[tid] = best; ri[tid] = bi;
        __syncthreads();
        for (int s = T >> 1; s > 0; s >>= 1) {
            if (tid < s) {
                if (rv[tid + s] < rv[tid] ||
                    (rv[tid + s] == rv[tid] && ri[tid + s] < ri[tid])) {
                    rv[tid] = rv[tid + s]; ri[tid] = ri[tid + s];
                }
            }
            __syncthreads();
        }
        if (tid == 0) {
            outIdx[(size_t)bs * k + j] = ri[0];
            if (outDist) outDist[(size_t)bs * k + j] = rv[0];
            dist[ri[0]] = 3.6e38f;
        }
        __syncthreads();
    }
}

// ---------------------------------------------------------------------------
// One conv1x1+BN+ReLU layer, LDS-resident f16 activations, wave-per-tile WMMA.
// A: two 16B ds loads per K-step. B: one 32B global load (pre-packed weight).
// ---------------------------------------------------------------------------
template<int KN>
__device__ __forceinline__ void
lds_mlp_layer(const _Float16* __restrict__ Xl, int ldx,
              _Float16* __restrict__ Yl, int ldy,
              int Cout, LayerP lp)
{
    const int lane = threadIdx.x & 31;
    const int wave = threadIdx.x >> 5;
    const int nw   = blockDim.x >> 5;
    const int mt   = KN >> 4;
    const int nt   = Cout >> 4;
    const int total = mt * nt;
    const int Kp   = lp.Kp;
    const int hv   = lane >> 4;     // half-wave: 0 or 1
    const int l16  = lane & 15;

    for (int t = wave; t < total; t += nw) {
        const int m0   = (t % mt) << 4;
        const int n0   = (t / mt) << 4;
        const int arow = m0 + l16;
        const int col  = n0 + l16;
        const _Float16* xrow = Xl + arow * ldx + hv * 16;
        const _Float16* wrow = lp.Wp + (size_t)col * Kp + hv * 16;
        v8f acc = {0.f, 0.f, 0.f, 0.f, 0.f, 0.f, 0.f, 0.f};

        for (int kk = 0; kk < Kp; kk += 32) {
            union { v16h v; v8h p[2]; } A;
            A.p[0] = *(const v8h*)(xrow + kk);
            A.p[1] = *(const v8h*)(xrow + kk + 8);
            const v16h Bv = *(const v16h*)(wrow + kk);
            acc = __builtin_amdgcn_wmma_f32_16x16x32_f16(
                false, A.v, false, Bv, (short)0, acc, false, false);
        }

        const float sc = lp.g[col] * rsqrtf(lp.var[col] + BN_EPS);
        const float sh = lp.bt[col] + (lp.b[col] - lp.mu[col]) * sc;
        const int mb = m0 + hv * 8;
        #pragma unroll
        for (int r = 0; r < 8; ++r) {
            float y = acc[r] * sc + sh;
            y = y > 0.f ? y : 0.f;
            Yl[(mb + r) * ldy + col] = (_Float16)y;
        }
    }
}

// ---------------------------------------------------------------------------
// Fused set-abstraction: gather(rel xyz || prev feats) -> LDS f16 ->
// 3x (WMMA conv1x1+BN+ReLU) -> maxpool over KN neighbors. One WG per centroid.
// ---------------------------------------------------------------------------
template<int KN, int CIN, int C1, int C2, int C3, int LD0, int LD1>
__global__ void __launch_bounds__(256)
sa_mlp_kernel(const float* __restrict__ qXyz, const float* __restrict__ srcXyz,
              const int* __restrict__ knnIdx, const _Float16* __restrict__ prevPts,
              int S, int Nsrc, Mlp3 mp, _Float16* __restrict__ outPts)
{
    __shared__ alignas(32) _Float16 X0[KN * LD0];
    __shared__ alignas(32) _Float16 X1[KN * LD1];

    const int bs  = blockIdx.x;
    const int b   = bs / S;
    const int tid = threadIdx.x;
    constexpr int CINP = (CIN + 31) & ~31;

    const float qx = qXyz[(size_t)bs * 3 + 0];
    const float qy = qXyz[(size_t)bs * 3 + 1];
    const float qz = qXyz[(size_t)bs * 3 + 2];

    if (tid < KN) {
        const int id = knnIdx[(size_t)bs * KN + tid];
        const float* p = srcXyz + ((size_t)b * Nsrc + id) * 3;
        _Float16* row = X0 + tid * LD0;
        row[0] = (_Float16)(p[0] - qx);
        row[1] = (_Float16)(p[1] - qy);
        row[2] = (_Float16)(p[2] - qz);
        if (CIN > 3) {
            const _Float16* f = prevPts + ((size_t)b * Nsrc + id) * (CIN - 3);
            for (int c = 0; c < CIN - 3; ++c) row[3 + c] = f[c];
        }
        for (int c = CIN; c < CINP; ++c) row[c] = (_Float16)0.f;   // zero-pad K
    }
    __syncthreads();

    lds_mlp_layer<KN>(X0, LD0, X1, LD1, C1, mp.l[0]);
    __syncthreads();
    lds_mlp_layer<KN>(X1, LD1, X0, LD0, C2, mp.l[1]);
    __syncthreads();
    lds_mlp_layer<KN>(X0, LD0, X1, LD1, C3, mp.l[2]);
    __syncthreads();

    for (int c = tid; c < C3; c += blockDim.x) {
        float m = -3.4e38f;
        for (int j = 0; j < KN; ++j) {
            float v = (float)X1[j * LD1 + c];
            m = v > m ? v : m;
        }
        outPts[(size_t)bs * C3 + c] = (_Float16)m;
    }
}

// ---------------------------------------------------------------------------
// Generic WMMA GEMM + BN + ReLU, 16x64 macro-tile per wave (4 N-tiles).
// X: f16 [M, Kp] (zero-padded, stride Kp), Y: f16 [M, C], C % 64 == 0.
// Per K-step: 1 A-fragment (2x b128), 4 B-fragments (8x b128),
// 4 independent v_wmma chains. Wave-uniform, guard-free.
// ---------------------------------------------------------------------------
__global__ void __launch_bounds__(256)
gemm_bn_relu_kernel(const _Float16* __restrict__ X, _Float16* __restrict__ Y,
                    int M, int C, LayerP lp)
{
    const int lane = threadIdx.x & 31;
    const int wave = threadIdx.x >> 5;
    const int nw   = blockDim.x >> 5;
    const int gw   = blockIdx.x * nw + wave;
    const int gs   = gridDim.x * nw;
    const int mt   = M >> 4;
    const int nt4  = C >> 6;            // macro N-tiles of 64 columns
    const int total = mt * nt4;
    const int Kp   = lp.Kp;
    const int hv   = lane >> 4;
    const int l16  = lane & 15;

    for (int t = gw; t < total; t += gs) {
        const int m0      = (t % mt) << 4;
        const int n0      = (t / mt) << 6;
        const int arow    = m0 + l16;
        const int colbase = n0 + l16;
        const _Float16* xrow = X + (size_t)arow * Kp + hv * 16;
        const _Float16* wrow[4];
        #pragma unroll
        for (int j = 0; j < 4; ++j)
            wrow[j] = lp.Wp + (size_t)(colbase + j * 16) * Kp + hv * 16;

        v8f acc[4];
        #pragma unroll
        for (int j = 0; j < 4; ++j)
            acc[j] = (v8f){0.f, 0.f, 0.f, 0.f, 0.f, 0.f, 0.f, 0.f};

        for (int kk = 0; kk < Kp; kk += 32) {
            union { v16h v; v8h p[2]; } A;
            A.p[0] = *(const v8h*)(xrow + kk);
            A.p[1] = *(const v8h*)(xrow + kk + 8);
            if (kk + 32 < Kp)    // uniform, speculative
                __builtin_prefetch(xrow + kk + 32, 0, 1);
            #pragma unroll
            for (int j = 0; j < 4; ++j) {
                const v16h Bv = *(const v16h*)(wrow[j] + kk);
                acc[j] = __builtin_amdgcn_wmma_f32_16x16x32_f16(
                    false, A.v, false, Bv, (short)0, acc[j], false, false);
            }
        }

        const int mb = m0 + hv * 8;
        #pragma unroll
        for (int j = 0; j < 4; ++j) {
            const int col = colbase + j * 16;
            const float sc = lp.g[col] * rsqrtf(lp.var[col] + BN_EPS);
            const float sh = lp.bt[col] + (lp.b[col] - lp.mu[col]) * sc;
            #pragma unroll
            for (int r = 0; r < 8; ++r) {
                float y = acc[j][r] * sc + sh;
                Y[(size_t)(mb + r) * C + col] = (_Float16)(y > 0.f ? y : 0.f);
            }
        }
    }
}

// ---------------------------------------------------------------------------
// SA3 input build: mean-center l2_xyz and concat l2 feats -> f16 [B*128, 288].
// ---------------------------------------------------------------------------
__global__ void __launch_bounds__(128)
sa3_build_kernel(const float* __restrict__ l2xyz, const _Float16* __restrict__ l2pts,
                 _Float16* __restrict__ out)
{
    const int b   = blockIdx.x;
    const int tid = threadIdx.x;
    __shared__ float mean3[3];
    if (tid < 3) {
        float s = 0.f;
        for (int i = 0; i < 128; ++i) s += l2xyz[((size_t)b * 128 + i) * 3 + tid];
        mean3[tid] = s * (1.f / 128.f);
    }
    __syncthreads();
    if (tid < 128) {
        _Float16* row = out + ((size_t)b * 128 + tid) * 288;
        const float* p = l2xyz + ((size_t)b * 128 + tid) * 3;
        row[0] = (_Float16)(p[0] - mean3[0]);
        row[1] = (_Float16)(p[1] - mean3[1]);
        row[2] = (_Float16)(p[2] - mean3[2]);
        const _Float16* f = l2pts + ((size_t)b * 128 + tid) * 256;
        for (int c = 0; c < 256; ++c) row[3 + c] = f[c];
        for (int c = 259; c < 288; ++c) row[c] = (_Float16)0.f;
    }
}

// Max over rowsPerBatch rows per batch of a f16 [B*rows, C] tensor -> [B, C].
__global__ void maxpool_rows_kernel(const _Float16* __restrict__ in,
                                    _Float16* __restrict__ out,
                                    int rowsPerBatch, int C, int Btot)
{
    const int gId = blockIdx.x * blockDim.x + threadIdx.x;
    if (gId >= Btot * C) return;
    const int b = gId / C, c = gId % C;
    float m = -3.4e38f;
    for (int r = 0; r < rowsPerBatch; ++r) {
        float v = (float)in[((size_t)b * rowsPerBatch + r) * C + c];
        m = v > m ? v : m;
    }
    out[gId] = (_Float16)m;
}

// FP3: l3 has a single point => interpolation is a pure broadcast.
__global__ void __launch_bounds__(256)
fp3_build_kernel(const _Float16* __restrict__ l2pts, const _Float16* __restrict__ l3feat,
                 _Float16* __restrict__ out)
{
    const int r = blockIdx.x;        // 0 .. B*128-1
    const int b = r >> 7;
    for (int c = threadIdx.x; c < 1280; c += blockDim.x) {
        _Float16 v = (c < 256) ? l2pts[(size_t)r * 256 + c]
                               : l3feat[(size_t)b * 1024 + (c - 256)];
        out[(size_t)r * 1280 + c] = v;
    }
}

// 3-NN inverse-distance interp + concat -> f16 [rows, Cpad] (zero-padded).
template<typename OwnT>
__global__ void __launch_bounds__(256)
interp_concat_kernel(const OwnT* __restrict__ own, int ownC,
                     const _Float16* __restrict__ feats, int featC, int featRowsPerB,
                     const int* __restrict__ idx3, const float* __restrict__ d3,
                     int rowsPerB, int Cpad, _Float16* __restrict__ out)
{
    const int r = blockIdx.x;
    const int b = r / rowsPerB;
    const float r0 = 1.f / (d3[(size_t)r * 3 + 0] + 1e-8f);
    const float r1 = 1.f / (d3[(size_t)r * 3 + 1] + 1e-8f);
    const float r2 = 1.f / (d3[(size_t)r * 3 + 2] + 1e-8f);
    const float inv = 1.f / (r0 + r1 + r2);
    const float w0 = r0 * inv, w1 = r1 * inv, w2 = r2 * inv;
    const _Float16* f0 = feats + ((size_t)b * featRowsPerB + idx3[(size_t)r * 3 + 0]) * featC;
    const _Float16* f1 = feats + ((size_t)b * featRowsPerB + idx3[(size_t)r * 3 + 1]) * featC;
    const _Float16* f2 = feats + ((size_t)b * featRowsPerB + idx3[(size_t)r * 3 + 2]) * featC;
    const int C = ownC + featC;
    for (int c = threadIdx.x; c < Cpad; c += blockDim.x) {
        float v = 0.f;
        if (c < ownC) v = (float)own[(size_t)r * ownC + c];
        else if (c < C) {
            const int cc = c - ownC;
            v = w0 * (float)f0[cc] + w1 * (float)f1[cc] + w2 * (float)f2[cc];
        }
        out[(size_t)r * Cpad + c] = (_Float16)v;
    }
}

// Final classifier head: out[r, 0:2] = feat[r, :] * W2 + b2 (f32 out).
__global__ void head_final_kernel(const _Float16* __restrict__ feat,
                                  const float* __restrict__ W2,
                                  const float* __restrict__ b2,
                                  float* __restrict__ out, int rows)
{
    const int r = blockIdx.x * blockDim.x + threadIdx.x;
    if (r >= rows) return;
    float a0 = b2[0], a1 = b2[1];
    const _Float16* f = feat + (size_t)r * 128;
    for (int c = 0; c < 128; ++c) {
        const float v = (float)f[c];
        a0 += v * W2[c * 2 + 0];
        a1 += v * W2[c * 2 + 1];
    }
    out[(size_t)r * 2 + 0] = a0;
    out[(size_t)r * 2 + 1] = a1;
}

// Two broadcast 3x3 identity matrices, [8,3,3] each.
__global__ void eye_kernel(float* __restrict__ out)
{
    const int t = blockIdx.x * blockDim.x + threadIdx.x;
    if (t >= 72) return;
    const int i = (t % 9) / 3, j = t % 3;
    const float v = (i == j) ? 1.f : 0.f;
    out[t] = v;
    out[72 + t] = v;
}

// ---------------------------------------------------------------------------
// Host launcher
// ---------------------------------------------------------------------------
static void launch_gemm(hipStream_t st, const _Float16* X, const LayerP& lp,
                        _Float16* Y, int M, int C)
{
    const int total = (M / 16) * (C / 64);   // 16x64 macro-tiles
    int blocks = (total + 7) / 8;
    if (blocks < 1) blocks = 1;
    if (blocks > 8192) blocks = 8192;
    gemm_bn_relu_kernel<<<blocks, 256, 0, st>>>(X, Y, M, C, lp);
}

extern "C" void kernel_launch(void* const* d_in, const int* in_sizes, int n_in,
                              void* d_out, int out_size, void* d_ws, size_t ws_size,
                              hipStream_t stream)
{
    (void)in_sizes; (void)n_in; (void)out_size; (void)ws_size;

    const float* x = (const float*)d_in[0];   // [8,4096,3]

    // ---- workspace carve (fixed, deterministic layout; 64B-aligned) ----
    char*  w   = (char*)d_ws;
    size_t off = 0;
    auto alignUp = [&](size_t a) { off = (off + a - 1) & ~(a - 1); };
    auto allocF = [&](size_t n) { alignUp(64); float* p = (float*)(w + off); off += n * 4; return p; };
    auto allocI = [&](size_t n) { alignUp(64); int*   p = (int*)(w + off);   off += n * 4; return p; };
    auto allocH = [&](size_t n) { alignUp(64); _Float16* p = (_Float16*)(w + off); off += n * 2; return p; };

    // Pack one conv+BN layer: transpose+pad weights to f16 on device.
    auto makeLayer = [&](int base, int K, int C) {
        const int Kp = (K + 31) & ~31;
        _Float16* Wp = allocH((size_t)C * Kp);
        pack_w_kernel<<<((C * Kp) + 255) / 256, 256, 0, stream>>>(
            (const float*)d_in[base], K, C, Kp, Wp);
        LayerP p{ Wp,
                  (const float*)d_in[base + 1], (const float*)d_in[base + 2],
                  (const float*)d_in[base + 3], (const float*)d_in[base + 4],
                  (const float*)d_in[base + 5], Kp };
        return p;
    };

    // Flattened pytree: x, sa1(3x6), sa2(3x6), sa3(3x6), fp3(2x6), fp2(2x6),
    // fp1(3x6), head(hd1 1x6 + W2 + b2).
    Mlp3 sa1{{ makeLayer(1, 3, 64),     makeLayer(7, 64, 64),    makeLayer(13, 64, 128) }};
    Mlp3 sa2{{ makeLayer(19, 131, 128), makeLayer(25, 128, 128), makeLayer(31, 128, 256) }};
    LayerP sa3a = makeLayer(37, 259, 256);
    LayerP sa3b = makeLayer(43, 256, 512);
    LayerP sa3c = makeLayer(49, 512, 1024);
    LayerP fp3a = makeLayer(55, 1280, 256);
    LayerP fp3b = makeLayer(61, 256, 256);
    LayerP fp2a = makeLayer(67, 384, 256);
    LayerP fp2b = makeLayer(73, 256, 128);
    LayerP fp1a = makeLayer(79, 131, 128);
    LayerP fp1b = makeLayer(85, 128, 128);
    LayerP fp1c = makeLayer(91, 128, 128);
    LayerP hd1  = makeLayer(97, 128, 128);
    const float* W2 = (const float*)d_in[103];  // [128,2]
    const float* b2 = (const float*)d_in[104];  // [2]

    float*     l1_xyz  = allocF((size_t)8 * 512 * 3);
    float*     l2_xyz  = allocF((size_t)8 * 128 * 3);
    _Float16*  l1_pts  = allocH((size_t)8 * 512 * 128);
    _Float16*  l2_pts  = allocH((size_t)8 * 128 * 256);
    _Float16*  l3_feat = allocH((size_t)8 * 1024);
    _Float16*  fp3_out = allocH((size_t)8 * 128 * 256);
    _Float16*  fp2_out = allocH((size_t)8 * 512 * 128);
    int*   idx1  = allocI((size_t)8 * 512 * 32);
    int*   idx2  = allocI((size_t)8 * 128 * 64);
    int*   idx3a = allocI((size_t)8 * 512 * 3);
    float* d3a   = allocF((size_t)8 * 512 * 3);
    int*   idx3b = allocI((size_t)8 * 4096 * 3);
    float* d3b   = allocF((size_t)8 * 4096 * 3);
    _Float16* bufA = allocH((size_t)32768 * 160);   // max: FP1 input [32768,160]
    _Float16* bufB = allocH((size_t)32768 * 160);

    // ---- SA1: FPS 4096->512, kNN32, MLP [3->64->64->128], maxpool ----
    fps_kernel<<<8, 256, 0, stream>>>(x, 4096, 512, l1_xyz);
    knn_kernel<<<8 * 512, 128, 0, stream>>>(l1_xyz, x, 512, 4096, 32, idx1, nullptr);
    sa_mlp_kernel<32, 3, 64, 64, 128, 72, 136><<<8 * 512, 256, 0, stream>>>(
        l1_xyz, x, idx1, nullptr, 512, 4096, sa1, l1_pts);

    // ---- SA2: FPS 512->128, kNN64, MLP [131->128->128->256], maxpool ----
    fps_kernel<<<8, 256, 0, stream>>>(l1_xyz, 512, 128, l2_xyz);
    knn_kernel<<<8 * 128, 128, 0, stream>>>(l2_xyz, l1_xyz, 128, 512, 64, idx2, nullptr);
    sa_mlp_kernel<64, 131, 128, 128, 256, 168, 264><<<8 * 128, 256, 0, stream>>>(
        l2_xyz, l1_xyz, idx2, l1_pts, 128, 512, sa2, l2_pts);

    // ---- SA3: global group, MLP [259->256->512->1024], maxpool over 128 ----
    sa3_build_kernel<<<8, 128, 0, stream>>>(l2_xyz, l2_pts, bufA);        // [1024,288]
    launch_gemm(stream, bufA, sa3a, bufB, 1024, 256);
    launch_gemm(stream, bufB, sa3b, bufA, 1024, 512);
    launch_gemm(stream, bufA, sa3c, bufB, 1024, 1024);
    maxpool_rows_kernel<<<(8 * 1024 + 255) / 256, 256, 0, stream>>>(bufB, l3_feat, 128, 1024, 8);

    // ---- FP3: broadcast-interp (M=1) + MLP [1280->256->256] ----
    fp3_build_kernel<<<1024, 256, 0, stream>>>(l2_pts, l3_feat, bufA);    // [1024,1280]
    launch_gemm(stream, bufA, fp3a, bufB, 1024, 256);
    launch_gemm(stream, bufB, fp3b, fp3_out, 1024, 256);

    // ---- FP2: 3NN interp onto l1 points + MLP [384->256->128] ----
    knn_kernel<<<8 * 512, 128, 0, stream>>>(l1_xyz, l2_xyz, 512, 128, 3, idx3a, d3a);
    interp_concat_kernel<<<8 * 512, 256, 0, stream>>>(l1_pts, 128, fp3_out, 256, 128,
                                                      idx3a, d3a, 512, 384, bufA);
    launch_gemm(stream, bufA, fp2a, bufB, 4096, 256);
    launch_gemm(stream, bufB, fp2b, fp2_out, 4096, 128);

    // ---- FP1: 3NN interp onto all points + MLP [131->128->128->128] ----
    knn_kernel<<<8 * 4096, 128, 0, stream>>>(x, l1_xyz, 4096, 512, 3, idx3b, d3b);
    interp_concat_kernel<<<8 * 4096, 256, 0, stream>>>(x, 3, fp2_out, 128, 512,
                                                       idx3b, d3b, 4096, 160, bufA);
    launch_gemm(stream, bufA, fp1a, bufB, 32768, 128);
    launch_gemm(stream, bufB, fp1b, bufA, 32768, 128);
    launch_gemm(stream, bufA, fp1c, bufB, 32768, 128);

    // ---- Head: conv1+bn1+relu (WMMA), then 128->2 linear ----
    launch_gemm(stream, bufB, hd1, bufA, 32768, 128);
    head_final_kernel<<<(32768 + 255) / 256, 256, 0, stream>>>(bufA, W2, b2,
                                                               (float*)d_out, 32768);
    eye_kernel<<<1, 128, 0, stream>>>((float*)d_out + (size_t)8 * 4096 * 2);
}
